// RelMultiHeadAttention_73151882985864
// MI455X (gfx1250) — matrix operand order repaired and checked by reading, not compiled, not run on hardware
//
#include <hip/hip_runtime.h>

// ---------------------------------------------------------------------------
// RelMultiHeadAttention for MI455X (gfx1250, wave32)
// WMMA f32_16x16x32_f16 compute + Tensor Data Mover (TDM) tile staging.
// B=8, L=512, D=1024, H=16, HD=64
// ---------------------------------------------------------------------------

typedef _Float16 half_t;
typedef __attribute__((ext_vector_type(8)))  _Float16 v8h;
typedef __attribute__((ext_vector_type(16))) _Float16 v16h;
typedef __attribute__((ext_vector_type(8)))  float    v8f;
typedef __attribute__((ext_vector_type(4)))  unsigned int v4u;
typedef __attribute__((ext_vector_type(8)))  int      v8i;
typedef __attribute__((ext_vector_type(4)))  int      v4i;

union V16 { v16h v; v8h h8[2]; _Float16 h[16]; };

#define WMMA_F32_F16(a, b, c) \
  __builtin_amdgcn_wmma_f32_16x16x32_f16(false, (a), false, (b), (short)0, (c), false, false)

constexpr int kB  = 8;
constexpr int kL  = 512;
constexpr int kD  = 1024;
constexpr int kH  = 16;
constexpr int kHD = 64;

// ---------------------------------------------------------------------------
// TDM: DMA a dense 2-D f16 tile (16 DWORDs wide x `rows` tall) from global to
// LDS with hardware padding (+8 DW after every 16 DW) => 48-half row stride.
// D# built per CDNA5 ISA 8.3/8.4; issued by one wave, tracked on TENSORcnt.
// Toolchain here is the 6-arg builtin: (v4u, v8i, v4i, v4i, v8i, cpol).
// ---------------------------------------------------------------------------
__device__ __forceinline__ void tdm_load_tile(const half_t* gptr,
                                              unsigned lds_byte_off,
                                              unsigned tensor_dw0,
                                              unsigned rows) {
  unsigned long long ga = (unsigned long long)(uintptr_t)gptr;
  // group 0: count=1 (valid user desc), lds_addr, 57-bit global addr, type=2
  v4u g0 = { 1u,
             lds_byte_off,
             (unsigned)ga,
             (unsigned)((ga >> 32) & 0x1FFFFFFu) | (2u << 30) };
  // group 1: data_size=4B, pad_enable, pad_interval=16DW(3), pad_amount=8DW(7)
  v8i g1;
  g1[0] = (int)((2u << 16) | (1u << 20) | (3u << 22) | (7u << 25));
  g1[1] = (int)((tensor_dw0 & 0xFFFFu) << 16);                  // tensor_dim0 lo
  g1[2] = (int)((tensor_dw0 >> 16) | ((rows & 0xFFFFu) << 16)); // dim0 hi|dim1 lo
  g1[3] = (int)((rows >> 16) | (16u << 16));                    // dim1 hi | tile_dim0=16
  g1[4] = (int)(rows & 0xFFFFu);                                // tile_dim1=rows, tile_dim2=0
  g1[5] = (int)tensor_dw0;                                      // tensor_dim0_stride lo32
  g1[6] = 0;                                                    // stride hi | dim1_stride lo
  g1[7] = 0;
  v4i g2 = {0, 0, 0, 0};                                        // 2-D tensor: unused
  v4i g3 = {0, 0, 0, 0};
  v8i g4 = {0, 0, 0, 0, 0, 0, 0, 0};                            // trailing group (unused)
  __builtin_amdgcn_tensor_load_to_lds(g0, g1, g2, g3, g4, 0);
}

__device__ __forceinline__ unsigned lds_off32(const void* p) {
  // generic (flat) pointer into LDS aperture: low 32 bits are the LDS offset
  return (unsigned)(uintptr_t)p;
}

// ---------------------------------------------------------------------------
// fp32 -> fp16 conversion (one-time; data then lives in the 192MB L2)
// ---------------------------------------------------------------------------
__global__ void cvt_f32_to_f16(const float* __restrict__ in,
                               half_t* __restrict__ out, int n) {
  int i = blockIdx.x * blockDim.x + threadIdx.x;
  if (i < n) out[i] = (half_t)in[i];
}

// ---------------------------------------------------------------------------
// RoPE (interleaved rotate-half) + head split (B,L,D)->(B,H,L,HD), f16
// scale folds the 1/sqrt(HD) attention scale into Q.
// ---------------------------------------------------------------------------
__global__ void rope_split(const half_t* __restrict__ in,
                           const float* __restrict__ sin_enc,
                           half_t* __restrict__ out, int do_rope, float scale) {
  int p = blockIdx.x * blockDim.x + threadIdx.x;       // pair index
  if (p >= kB * kL * (kD / 2)) return;
  int j  = p & (kD / 2 - 1);
  int l  = (p >> 9) & (kL - 1);
  int b  = p >> 18;
  int d0 = j * 2;
  float x0 = (float)in[((size_t)(b * kL + l)) * kD + d0];
  float x1 = (float)in[((size_t)(b * kL + l)) * kD + d0 + 1];
  float y0 = x0, y1 = x1;
  if (do_rope) {
    float s = sin_enc[(size_t)l * kD + d0];       // sin_pos (even col)
    float c = sin_enc[(size_t)l * kD + d0 + 1];   // cos_pos (odd col)
    y0 = x0 * c - x1 * s;
    y1 = x1 * c + x0 * s;
  }
  y0 *= scale; y1 *= scale;
  int h  = d0 >> 6;
  int hd = d0 & 63;
  half_t* o = out + ((((size_t)(b * kH + h)) * kL + l) * kHD + hd);
  o[0] = (half_t)y0;
  o[1] = (half_t)y1;
}

// ---------------------------------------------------------------------------
// GEMM: C[M,N] = A[M,K] @ W[N,K]^T + bias[N]     (A, W f16; accum f32)
// 128x128x32 block tile, 8 waves, each wave a 32x64 WMMA sub-tile.
// TDM double-buffered tile staging: wave 0 drives the DMA engine while all
// waves stream WMMAs from the live LDS buffer.
// ---------------------------------------------------------------------------
template <typename OutT>
__global__ __launch_bounds__(256) void gemm_xwt(
    const half_t* __restrict__ A, const half_t* __restrict__ W,
    const float* __restrict__ bias, OutT* __restrict__ Cout,
    int M, int N, int K) {
  constexpr int BM = 128, BN = 128, BK = 32;
  constexpr int LDT = BK + 16;                 // 48 halves (TDM pad: 16DW+8DW)
  __shared__ half_t As[2][BM * LDT];
  __shared__ half_t Bs[2][BN * LDT];

  const int tid  = threadIdx.x;
  const int wave = tid >> 5;
  const int lane = tid & 31;
  const int nl   = lane & 15;
  const int kh   = lane >> 4;

  const int bm = blockIdx.y * BM;
  const int bn = blockIdx.x * BN;
  const int wm = (wave & 3) * 32;
  const int wn = (wave >> 2) * 64;

  v8f acc[2][4];
  for (int i = 0; i < 2; ++i)
    for (int j = 0; j < 4; ++j)
      for (int e = 0; e < 8; ++e) acc[i][j][e] = 0.f;

  const int nkt = K / BK;
  const unsigned kdw = (unsigned)(K / 2);      // row length in DWORDs

  auto issueTiles = [&](int kt, int buf) {
    if (wave == 0) {
      tdm_load_tile(A + (size_t)bm * K + kt * BK, lds_off32(&As[buf][0]), kdw, BM);
      tdm_load_tile(W + (size_t)bn * K + kt * BK, lds_off32(&Bs[buf][0]), kdw, BN);
    }
  };

  issueTiles(0, 0);
  if (wave == 0) __builtin_amdgcn_s_wait_tensorcnt(0);
  __syncthreads();

  int cur = 0;
  for (int kt = 0; kt < nkt; ++kt) {
    if (kt + 1 < nkt) issueTiles(kt + 1, cur ^ 1);   // DMA next tile

    V16 afr[2], bfr[4];
#pragma unroll
    for (int i = 0; i < 2; ++i) {         // A frag: lane = M, K split by half
      const half_t* p = &As[cur][0] + (wm + i * 16 + nl) * LDT;
      afr[i].h8[0] = *(const v8h*)(p + kh * 8);
      afr[i].h8[1] = *(const v8h*)(p + 16 + kh * 8);
    }
#pragma unroll
    for (int j = 0; j < 4; ++j) {         // B frag: lane = N, K split by half
      const half_t* p = &Bs[cur][0] + (wn + j * 16 + nl) * LDT + kh * 8;
      bfr[j].h8[0] = *(const v8h*)(p);
      bfr[j].h8[1] = *(const v8h*)(p + 16);
    }
#pragma unroll
    for (int i = 0; i < 2; ++i)
#pragma unroll
      for (int j = 0; j < 4; ++j)
        acc[i][j] = WMMA_F32_F16(afr[i].v, bfr[j].v, acc[i][j]);

    if (wave == 0 && kt + 1 < nkt) __builtin_amdgcn_s_wait_tensorcnt(0);
    __syncthreads();                       // publish next buffer / retire reads
    cur ^= 1;
  }

#pragma unroll
  for (int i = 0; i < 2; ++i)
#pragma unroll
    for (int j = 0; j < 4; ++j)
#pragma unroll
      for (int r = 0; r < 8; ++r) {       // C frag: lane = N, 8 rows per lane
        int row = bm + wm + i * 16 + r + kh * 8;
        int col = bn + wn + j * 16 + nl;
        float vv = acc[i][j][r] + bias[col];
        Cout[(size_t)row * N + col] = (OutT)vv;
      }
}

// ---------------------------------------------------------------------------
// Fused flash attention with distance bias + key padding mask.
// Grid (qtile=8, h=16, b=8); 128 threads = 4 waves; each wave owns 16 q-rows.
// Scores never touch HBM: online softmax in registers, P transposed via LDS.
// ---------------------------------------------------------------------------
__global__ __launch_bounds__(128) void flash_attn(
    const half_t* __restrict__ qh, const half_t* __restrict__ khd,
    const half_t* __restrict__ vhd, const int* __restrict__ dist,
    const float* __restrict__ dist_emb, const int* __restrict__ length,
    half_t* __restrict__ ctx) {
  constexpr int LT = 72;                  // padded half-stride (144B rows)
  __shared__ half_t Qs[64 * LT];
  __shared__ half_t Ks[64 * LT];
  __shared__ half_t Vt[64 * LT];          // V transposed: [d][key]
  __shared__ half_t Ps[4][16 * LT];       // per-wave P staging (C->A relayout)
  __shared__ float  emb_sh[64];

  const int tid  = threadIdx.x;
  const int wave = tid >> 5;
  const int lane = tid & 31;
  const int nl   = lane & 15;
  const int kh2  = lane >> 4;
  const int qt = blockIdx.x, h = blockIdx.y, b = blockIdx.z;
  const int qbase = qt * 64;
  const int len = length[b];

  if (tid < 64) emb_sh[tid] = dist_emb[tid * kH + h];

  const size_t headoff = ((size_t)(b * kH + h)) * kL * kHD;

  // load Q tile (64 x 64 f16)
#pragma unroll
  for (int i = 0; i < 2; ++i) {
    int c = tid * 2 + i;
    int row = c >> 2, col = (c & 3) * 8;
    *(uint4*)(Qs + row * LT + col) =
        *(const uint4*)(qh + headoff + (size_t)(qbase + row) * kHD + col);
  }
  __syncthreads();

  V16 aq[2];
#pragma unroll
  for (int f = 0; f < 2; ++f) {
    const half_t* p = Qs + (wave * 16 + nl) * LT + f * 32;
    aq[f].h8[0] = *(const v8h*)(p + kh2 * 8);
    aq[f].h8[1] = *(const v8h*)(p + 16 + kh2 * 8);
  }

  v8f accO[4];
  for (int j = 0; j < 4; ++j)
    for (int e = 0; e < 8; ++e) accO[j][e] = 0.f;
  float mrow[8], lrow[8];
  for (int r = 0; r < 8; ++r) { mrow[r] = -3.0e38f; lrow[r] = 0.f; }

  for (int kt = 0; kt < kL / 64; ++kt) {
    const int kb = kt * 64;
    __syncthreads();                      // protect Ks/Vt/Ps from prev iter
    // load K tile; load V tile and store transposed
#pragma unroll
    for (int i = 0; i < 2; ++i) {
      int c = tid * 2 + i;
      int row = c >> 2, col = (c & 3) * 8;
      *(uint4*)(Ks + row * LT + col) =
          *(const uint4*)(khd + headoff + (size_t)(kb + row) * kHD + col);
      uint4 vv = *(const uint4*)(vhd + headoff + (size_t)(kb + row) * kHD + col);
      const half_t* vhh = (const half_t*)&vv;
#pragma unroll
      for (int e = 0; e < 8; ++e) Vt[(col + e) * LT + row] = vhh[e];
    }
    __syncthreads();

    // S = Q K^T (16 q-rows x 64 keys per wave), f32 accum
    v8f s[4];
#pragma unroll
    for (int j = 0; j < 4; ++j) {
      for (int e = 0; e < 8; ++e) s[j][e] = 0.f;
#pragma unroll
      for (int f = 0; f < 2; ++f) {
        V16 bk;
        const half_t* p = Ks + (j * 16 + nl) * LT + f * 32 + kh2 * 8;
        bk.h8[0] = *(const v8h*)(p);
        bk.h8[1] = *(const v8h*)(p + 16);
        s[j] = WMMA_F32_F16(aq[f].v, bk.v, s[j]);
      }
    }

    // + dist_emb[dist[b,q,k]][h], key padding mask
#pragma unroll
    for (int j = 0; j < 4; ++j) {
      int key = kb + j * 16 + nl;
      bool oob = key >= len;
#pragma unroll
      for (int r = 0; r < 8; ++r) {
        int qrow = qbase + wave * 16 + r + kh2 * 8;
        int di = dist[((size_t)b * kL + qrow) * kL + key];
        float sv = s[j][r] + emb_sh[di];
        s[j][r] = oob ? -1e9f : sv;
      }
    }

    // online softmax (row stats live in the 16-lane group owning the row)
    float corr[8];
#pragma unroll
    for (int r = 0; r < 8; ++r) {
      float mx = fmaxf(fmaxf(s[0][r], s[1][r]), fmaxf(s[2][r], s[3][r]));
      mx = fmaxf(mx, __shfl_xor(mx, 1));
      mx = fmaxf(mx, __shfl_xor(mx, 2));
      mx = fmaxf(mx, __shfl_xor(mx, 4));
      mx = fmaxf(mx, __shfl_xor(mx, 8));
      float mnew = fmaxf(mrow[r], mx);
      float ps = 0.f;
#pragma unroll
      for (int j = 0; j < 4; ++j) {
        float pv = __expf(s[j][r] - mnew);
        s[j][r] = pv;
        ps += pv;
      }
      ps += __shfl_xor(ps, 1);
      ps += __shfl_xor(ps, 2);
      ps += __shfl_xor(ps, 4);
      ps += __shfl_xor(ps, 8);
      corr[r] = __expf(mrow[r] - mnew);
      lrow[r] = lrow[r] * corr[r] + ps;
      mrow[r] = mnew;
    }
#pragma unroll
    for (int j = 0; j < 4; ++j)
#pragma unroll
      for (int r = 0; r < 8; ++r) accO[j][r] *= corr[r];

    // stage P (C layout -> A layout) through per-wave LDS
#pragma unroll
    for (int j = 0; j < 4; ++j)
#pragma unroll
      for (int r = 0; r < 8; ++r)
        Ps[wave][(r + kh2 * 8) * LT + j * 16 + nl] = (half_t)s[j][r];
    __syncthreads();

    V16 ap[2];
#pragma unroll
    for (int f = 0; f < 2; ++f) {
      const half_t* p = Ps[wave] + nl * LT + f * 32;
      ap[f].h8[0] = *(const v8h*)(p + kh2 * 8);
      ap[f].h8[1] = *(const v8h*)(p + 16 + kh2 * 8);
    }
    // O += P V  (V read transposed so B-frags are two contiguous 16B runs)
#pragma unroll
    for (int j = 0; j < 4; ++j) {
#pragma unroll
      for (int f = 0; f < 2; ++f) {
        V16 bv;
        const half_t* p = Vt + (j * 16 + nl) * LT + f * 32 + kh2 * 8;
        bv.h8[0] = *(const v8h*)(p);
        bv.h8[1] = *(const v8h*)(p + 16);
        accO[j] = WMMA_F32_F16(ap[f].v, bv.v, accO[j]);
      }
    }
  }

  // normalize and write ctx in (B,L,D) f16 (head-merge)
#pragma unroll
  for (int j = 0; j < 4; ++j)
#pragma unroll
    for (int r = 0; r < 8; ++r) {
      int qrow = qbase + wave * 16 + r + kh2 * 8;
      float vv = accO[j][r] / lrow[r];
      ctx[((size_t)(b * kL) + qrow) * kD + h * kHD + j * 16 + nl] = (half_t)vv;
    }
}

// ---------------------------------------------------------------------------
// host-side orchestration
// ---------------------------------------------------------------------------
extern "C" void kernel_launch(void* const* d_in, const int* in_sizes, int n_in,
                              void* d_out, int out_size, void* d_ws, size_t ws_size,
                              hipStream_t stream) {
  (void)in_sizes; (void)n_in; (void)out_size; (void)ws_size;
  const float* q        = (const float*)d_in[0];
  const float* k        = (const float*)d_in[1];
  const float* v        = (const float*)d_in[2];
  const float* Wq       = (const float*)d_in[3];
  const float* bq       = (const float*)d_in[4];
  const float* Wk       = (const float*)d_in[5];
  const float* bk       = (const float*)d_in[6];
  const float* Wv       = (const float*)d_in[7];
  const float* bv       = (const float*)d_in[8];
  const float* Wo       = (const float*)d_in[9];
  const float* bo       = (const float*)d_in[10];
  const float* dist_emb = (const float*)d_in[11];
  const float* sin_enc  = (const float*)d_in[12];
  const int*   length   = (const int*)d_in[13];
  const int*   dist     = (const int*)d_in[14];
  float* out = (float*)d_out;

  char* ws = (char*)d_ws;
  const size_t ACT = (size_t)kB * kL * kD * sizeof(half_t);  // 8 MiB
  half_t* r0   = (half_t*)(ws + 0 * ACT);   // q16 -> later qh (B,H,L,HD)
  half_t* r1   = (half_t*)(ws + 1 * ACT);   // k16 -> later kh
  half_t* r2   = (half_t*)(ws + 2 * ACT);   // v16 -> later vh
  half_t* r3   = (half_t*)(ws + 3 * ACT);   // projection scratch (B,L,D)
  half_t* ctx  = (half_t*)(ws + 4 * ACT);   // attention output (B,L,D)
  half_t* wq16 = (half_t*)(ws + 5 * ACT);
  half_t* wk16 = wq16 + (size_t)kD * kD;
  half_t* wv16 = wk16 + (size_t)kD * kD;
  half_t* wo16 = wv16 + (size_t)kD * kD;

  const int nAct = kB * kL * kD;            // 4,194,304
  const int nWt  = kD * kD;                 // 1,048,576
  const int CT = 256;
  cvt_f32_to_f16<<<(nAct + CT - 1) / CT, CT, 0, stream>>>(q, r0, nAct);
  cvt_f32_to_f16<<<(nAct + CT - 1) / CT, CT, 0, stream>>>(k, r1, nAct);
  cvt_f32_to_f16<<<(nAct + CT - 1) / CT, CT, 0, stream>>>(v, r2, nAct);
  cvt_f32_to_f16<<<(nWt + CT - 1) / CT, CT, 0, stream>>>(Wq, wq16, nWt);
  cvt_f32_to_f16<<<(nWt + CT - 1) / CT, CT, 0, stream>>>(Wk, wk16, nWt);
  cvt_f32_to_f16<<<(nWt + CT - 1) / CT, CT, 0, stream>>>(Wv, wv16, nWt);
  cvt_f32_to_f16<<<(nWt + CT - 1) / CT, CT, 0, stream>>>(Wo, wo16, nWt);

  const dim3 gGemm(kD / 128, (kB * kL) / 128);   // (8, 32)
  const int nPair = kB * kL * (kD / 2);
  const dim3 gRope((nPair + CT - 1) / CT);

  // Q projection -> RoPE + split + 1/sqrt(HD) scale
  gemm_xwt<half_t><<<gGemm, 256, 0, stream>>>(r0, wq16, bq, r3, kB * kL, kD, kD);
  rope_split<<<gRope, CT, 0, stream>>>(r3, sin_enc, r0, 1, 0.125f);
  // K projection -> RoPE + split
  gemm_xwt<half_t><<<gGemm, 256, 0, stream>>>(r1, wk16, bk, r3, kB * kL, kD, kD);
  rope_split<<<gRope, CT, 0, stream>>>(r3, sin_enc, r1, 1, 1.0f);
  // V projection -> split only
  gemm_xwt<half_t><<<gGemm, 256, 0, stream>>>(r2, wv16, bv, r3, kB * kL, kD, kD);
  rope_split<<<gRope, CT, 0, stream>>>(r3, sin_enc, r2, 0, 1.0f);

  // fused attention with dist bias + mask
  flash_attn<<<dim3(kL / 64, kH, kB), 128, 0, stream>>>(
      r0, r1, r2, dist, dist_emb, length, ctx);

  // output projection -> fp32 result
  gemm_xwt<float><<<gGemm, 256, 0, stream>>>(ctx, wo16, bo, out, kB * kL, kD, kD);
}
